// CenterPooling_24584392803056
// MI455X (gfx1250) — matrix operand order repaired and checked by reading, not compiled, not run on hardware
//
#include <hip/hip_runtime.h>
#include <math.h>

// ---------------------------------------------------------------------------
// CenterPooling on MI455X (gfx1250, wave32, WMMA + async global->LDS).
//
// Identity: cummax(rev) + cummax(fwd) along an axis == broadcast of the axis
// max. So out = rowmax_W(convBNReLU(x,w_up)) + colmax_H(convBNReLU(x,w_down));
// the 1GB conv outputs never hit HBM, only two 4MB max surfaces.
//
// Conv = implicit GEMM with v_wmma_f32_16x16x32_bf16 (fp32->bf16 operands,
// fp32 accumulate).
//  * Weights are pre-converted ONCE (wprep) to bf16 in the exact LDS layout
//    [co][tap][ci] (ci-contiguous); conv-side A staging is then pure b128
//    copies issued on the ASYNC global->LDS path (no VGPR round-trip),
//    overlapped with the B-tile gather and fenced with s_wait_asynccnt
//    before the barrier (CDNA5 ISA).
//  * Per 32-channel chunk, the input is staged as a RAW tile
//    rt[3 keep-lines][128 pooled + 2 halo][32 ci] (ci-contiguous), so every
//    tap's B-fragment is a shifted pair of ds_load_b128 — x is read from
//    global exactly once per chunk, not once per tap. Pooled halo is always
//    OOB => permanent zero strip. Mode-0 staging uses paired float2 loads.
//  * 9 unrolled taps x 4 WMMA (2x2 register blocking per wave) run between a
//    single pair of barriers per chunk.
// Pooling is fused in the epilogue: BN+ReLU, max of the wave's two pixel
// sub-tiles, 16-lane shuffle max, one u32 atomicMax per co row (exact:
// post-ReLU values are >= 0, so u32 bit order == float order).
// ---------------------------------------------------------------------------

typedef __bf16 bf16;
typedef bf16  v16bf __attribute__((ext_vector_type(16)));
typedef float v8f   __attribute__((ext_vector_type(8)));

#define Bn 32
#define Cn 256
#define Hn 128
#define Wn 128
#define CO_TILE 64
#define PIX_TILE 128
#define CISTEP 32
#define NPIX (Bn * Hn * Wn)          // 524288
#define ROWMAX_ELEMS (Bn * Cn * Hn)  // 1048576 u32
#define COLMAX_ELEMS (Bn * Cn * Wn)  // 1048576 u32
#define WBF_ELEMS (Cn * 9 * Cn)      // 589824 bf16 per branch

__global__ void fill0_kernel(unsigned int* __restrict__ p, int n) {
    int i = blockIdx.x * blockDim.x + threadIdx.x;
    if (i < n) p[i] = 0u;   // 0u == +0.0f: identity for max over ReLU outputs
}

__global__ void bn_prep_kernel(const float* __restrict__ g_up, const float* __restrict__ b_up,
                               const float* __restrict__ m_up, const float* __restrict__ v_up,
                               const float* __restrict__ g_dn, const float* __restrict__ b_dn,
                               const float* __restrict__ m_dn, const float* __restrict__ v_dn,
                               float* __restrict__ bnp) {
    int c = threadIdx.x;
    if (c < Cn) {
        float su = g_up[c] * rsqrtf(v_up[c] + 1e-5f);
        bnp[c]          = su;
        bnp[Cn + c]     = b_up[c] - m_up[c] * su;
        float sd = g_dn[c] * rsqrtf(v_dn[c] + 1e-5f);
        bnp[2 * Cn + c] = sd;
        bnp[3 * Cn + c] = b_dn[c] - m_dn[c] * sd;
    }
}

// OIHW fp32 -> [co][tap][ci] bf16 (ci-contiguous rows, matches LDS As9 layout)
__global__ void wprep_kernel(const float* __restrict__ w, bf16* __restrict__ out) {
    int gid = blockIdx.x * blockDim.x + threadIdx.x;    // = (co*9 + tap)*256 + ci
    if (gid < WBF_ELEMS) {
        int ci  = gid & 255;
        int t9  = gid >> 8;          // co*9 + tap
        int co  = t9 / 9;
        int tap = t9 - co * 9;
        out[gid] = (bf16)w[(size_t)co * (Cn * 9) + ci * 9 + tap];
    }
}

// mode 0 ("up"):   pixel order p = ((b*H)+h)*W + w -> pooled dim = W, keep = h
// mode 1 ("down"): pixel order p = ((b*W)+w)*H + h -> pooled dim = H, keep = w
__global__ __launch_bounds__(256)
void conv_wmma_kernel(const float* __restrict__ x, const bf16* __restrict__ wbf,
                      const float* __restrict__ scale, const float* __restrict__ shift,
                      unsigned int* __restrict__ red, int mode) {
    // A: all 9 taps, row = [tap*32 + ci] halves (576B rows, 16B-aligned)
    __shared__ __align__(16) bf16 As9[CO_TILE][9 * CISTEP];          // 36864 B
    // raw input tile: [keep-tap][pooled index 0..129][ci]  (64B ci-rows)
    __shared__ __align__(16) bf16 rt[3][PIX_TILE + 2][CISTEP];       // 24960 B

    const int tid    = threadIdx.x;
    const int lane   = tid & 31;
    const int waveId = tid >> 5;       // 8 waves
    const int coW    = waveId & 1;     // 2 co groups of 32
    const int pixW   = waveId >> 1;    // 4 pixel groups of 32
    const int lhalf  = lane & 15;
    const int hi     = lane >> 4;

    const int coBase = blockIdx.y * CO_TILE;
    const int p0     = blockIdx.x * PIX_TILE;       // tile = one full pooled line
    const int bb     = p0 >> 14;                    // batch (block-uniform)
    const int keep   = (p0 >> 7) & 127;             // h (mode0) / w (mode1)
    const float* xb  = x + ((size_t)bb << 22);      // b * C*H*W

    // A staging: thread -> (co = tid>>2, 8-ci slice at (tid&3)*8); 9 async b128 copies
    const int aCo   = tid >> 2;
    const int aPart = (tid & 3) * 8;
    const bf16* aBase = wbf + (size_t)(coBase + aCo) * (9 * Cn);

    // B staging: thread -> pooled pair (tid&63)*2; round r -> q = r*4 + (tid>>6):
    //            tk = q>>5, ci = q&31
    const int bPl2 = (tid & 63) * 2;
    const int bGrp = tid >> 6;

    // pooled-dim halo (coords -1 and 128) is always OOB: write zeros once
    if (tid < 192) {
        int ci = tid & 31, ppSel = (tid >> 5) & 1, tk = tid >> 6;
        rt[tk][ppSel ? (PIX_TILE + 1) : 0][ci] = (bf16)0.f;
    }

    v8f accs[2][2];
    #pragma unroll
    for (int cc = 0; cc < 2; ++cc)
        #pragma unroll
        for (int nn = 0; nn < 2; ++nn)
            accs[cc][nn] = (v8f){0.f, 0.f, 0.f, 0.f, 0.f, 0.f, 0.f, 0.f};

    for (int ci0 = 0; ci0 < Cn; ci0 += CISTEP) {
        // ---- stage A: async global->LDS b128 copies (ASYNCcnt path); the
        //      copies run in the background while we gather the B tile below.
        #pragma unroll
        for (int t = 0; t < 9; ++t) {
            const bf16* g = aBase + t * Cn + ci0 + aPart;
            // low 32 bits of a generic pointer to __shared__ == LDS offset
            unsigned l = (unsigned)(uintptr_t)&As9[aCo][t * CISTEP + aPart];
            asm volatile("global_load_async_to_lds_b128 %0, %1, off"
                         :: "v"(l), "v"(g)
                         : "memory");
        }

        // ---- stage raw input tile: 24 rounds, 2 pooled elems per round
        #pragma unroll 4
        for (int r = 0; r < 24; ++r) {
            const int q  = r * 4 + bGrp;
            const int tk = q >> 5;            // keep-dim tap 0..2
            const int ci = q & 31;
            const int kc = keep + tk - 1;     // keep-dim coordinate (uniform validity)
            float v0 = 0.f, v1 = 0.f;
            if ((unsigned)kc < 128u) {
                if (mode == 0) {              // pooled = w: contiguous pair
                    const float2 f2 = *(const float2*)(xb + (((size_t)(ci0 + ci)) << 14)
                                                          + (kc << 7) + bPl2);
                    v0 = f2.x; v1 = f2.y;
                } else {                      // pooled = h: stride-W column gather
                    const float* base = xb + (((size_t)(ci0 + ci)) << 14) + kc;
                    v0 = base[(size_t)bPl2 << 7];
                    v1 = base[(size_t)(bPl2 + 1) << 7];
                }
            }
            rt[tk][1 + bPl2][ci] = (bf16)v0;
            rt[tk][2 + bPl2][ci] = (bf16)v1;
        }

        // producer-side fence: this wave's async LDS writes must land before
        // the barrier makes them visible to consumers
        asm volatile("s_wait_asynccnt 0x0" ::: "memory");
        __syncthreads();

        // ---- 9 unrolled taps of WMMA from the staged tiles
        #pragma unroll
        for (int tap = 0; tap < 9; ++tap) {
            const int kh = tap / 3, kw = tap % 3;               // compile-time
            const int tk = (mode == 0) ? kh : kw;               // keep-dim tap
            const int tp = (mode == 0) ? kw : kh;               // pooled-dim tap

            union { uint4 q[2]; v16bf v; } a0, a1, b0, b1;
            // A 16x32: lanes 0-15 -> M=lane, K 0..7/16..23; lanes 16-31 -> K 8..15/24..31
            const int aM0 = coW * 32 + lhalf;
            a0.q[0] = *(const uint4*)&As9[aM0][tap * CISTEP + hi * 8];
            a0.q[1] = *(const uint4*)&As9[aM0][tap * CISTEP + 16 + hi * 8];
            a1.q[0] = *(const uint4*)&As9[aM0 + 16][tap * CISTEP + hi * 8];
            a1.q[1] = *(const uint4*)&As9[aM0 + 16][tap * CISTEP + 16 + hi * 8];
            // B 32x16: lanes 0-15 -> N=lane, K 0..7/16..23; lanes 16-31 -> N=lane-16, K 8..15/24..31
            // value for (pixel pl, tap) lives at rt[tk][pl + tp][ci]
            const int pl0 = pixW * 32 + lhalf + tp;
            b0.q[0] = *(const uint4*)&rt[tk][pl0][hi * 8];
            b0.q[1] = *(const uint4*)&rt[tk][pl0][16 + hi * 8];
            b1.q[0] = *(const uint4*)&rt[tk][pl0 + 16][hi * 8];
            b1.q[1] = *(const uint4*)&rt[tk][pl0 + 16][16 + hi * 8];

            accs[0][0] = __builtin_amdgcn_wmma_f32_16x16x32_bf16(false, a0.v, false, b0.v, (short)0, accs[0][0], false, false);
            accs[0][1] = __builtin_amdgcn_wmma_f32_16x16x32_bf16(false, a0.v, false, b1.v, (short)0, accs[0][1], false, false);
            accs[1][0] = __builtin_amdgcn_wmma_f32_16x16x32_bf16(false, a1.v, false, b0.v, (short)0, accs[1][0], false, false);
            accs[1][1] = __builtin_amdgcn_wmma_f32_16x16x32_bf16(false, a1.v, false, b1.v, (short)0, accs[1][1], false, false);
        }
        __syncthreads();
    }

    // ---- epilogue: BN + ReLU, pool both pixel sub-tiles + 16-lane max, atomic
    // every pixel in this block shares (bb, keep): one target entry per co.
    #pragma unroll
    for (int cc = 0; cc < 2; ++cc) {
        #pragma unroll
        for (int r = 0; r < 8; ++r) {
            // C/D layout: VGPR r holds M=r (lanes 0-15) and M=8+r (lanes 16-31)
            const int co = coBase + coW * 32 + cc * 16 + hi * 8 + r;
            const float s = scale[co], sh = shift[co];
            float v0 = fmaxf(accs[cc][0][r] * s + sh, 0.f);
            float v1 = fmaxf(accs[cc][1][r] * s + sh, 0.f);
            float v = fmaxf(v0, v1);
            #pragma unroll
            for (int m = 1; m < 16; m <<= 1)
                v = fmaxf(v, __shfl_xor(v, m, 16));   // max over 16-lane segment
            if (lhalf == 0) {
                int idx = (bb * Cn + co) * 128 + keep;    // Hn == Wn == 128
                atomicMax(&red[idx], __float_as_uint(v)); // v >= 0: bit order == float order
            }
        }
    }
}

__global__ void combine_kernel(const unsigned int* __restrict__ rowmax,
                               const unsigned int* __restrict__ colmax,
                               float4* __restrict__ out) {
    int gid = blockIdx.x * blockDim.x + threadIdx.x;   // float4 index, total 2^25
    int idx = gid << 2;
    int w = idx & 127;
    int h = (idx >> 7) & 127;
    int c = (idx >> 14) & 255;
    int b = idx >> 22;
    int base = b * Cn + c;
    float rv = __uint_as_float(rowmax[base * Hn + h]);
    const unsigned int* cm = &colmax[base * Wn + w];
    float4 o;
    o.x = rv + __uint_as_float(cm[0]);
    o.y = rv + __uint_as_float(cm[1]);
    o.z = rv + __uint_as_float(cm[2]);
    o.w = rv + __uint_as_float(cm[3]);
    out[gid] = o;
}

extern "C" void kernel_launch(void* const* d_in, const int* in_sizes, int n_in,
                              void* d_out, int out_size, void* d_ws, size_t ws_size,
                              hipStream_t stream) {
    const float* x    = (const float*)d_in[0];
    const float* w_up = (const float*)d_in[1];
    const float* g_up = (const float*)d_in[2];
    const float* b_up = (const float*)d_in[3];
    const float* m_up = (const float*)d_in[4];
    const float* v_up = (const float*)d_in[5];
    const float* w_dn = (const float*)d_in[6];
    const float* g_dn = (const float*)d_in[7];
    const float* b_dn = (const float*)d_in[8];
    const float* m_dn = (const float*)d_in[9];
    const float* v_dn = (const float*)d_in[10];

    unsigned int* ws     = (unsigned int*)d_ws;
    unsigned int* rowmax = ws;                                          // 4 MB
    unsigned int* colmax = ws + ROWMAX_ELEMS;                           // 4 MB
    float*        bnp    = (float*)(ws + ROWMAX_ELEMS + COLMAX_ELEMS);  // 4 KB
    bf16*         wbf_up = (bf16*)(bnp + 4 * Cn);                       // 1.125 MB
    bf16*         wbf_dn = wbf_up + WBF_ELEMS;                          // 1.125 MB

    const int redTotal = ROWMAX_ELEMS + COLMAX_ELEMS;
    fill0_kernel<<<(redTotal + 255) / 256, 256, 0, stream>>>(ws, redTotal);
    bn_prep_kernel<<<1, 256, 0, stream>>>(g_up, b_up, m_up, v_up,
                                          g_dn, b_dn, m_dn, v_dn, bnp);
    wprep_kernel<<<(WBF_ELEMS + 255) / 256, 256, 0, stream>>>(w_up, wbf_up);
    wprep_kernel<<<(WBF_ELEMS + 255) / 256, 256, 0, stream>>>(w_dn, wbf_dn);

    dim3 grid(NPIX / PIX_TILE, Cn / CO_TILE);   // (4096, 4)
    conv_wmma_kernel<<<grid, 256, 0, stream>>>(x, wbf_up, bnp,          bnp + Cn,     rowmax, 0);
    conv_wmma_kernel<<<grid, 256, 0, stream>>>(x, wbf_dn, bnp + 2 * Cn, bnp + 3 * Cn, colmax, 1);

    const int nVec4 = (Bn * Cn * Hn * Wn) / 4;  // 2^25
    combine_kernel<<<nVec4 / 256, 256, 0, stream>>>(rowmax, colmax, (float4*)d_out);
}
